// DynamicGCN_84318797955721
// MI455X (gfx1250) — matrix operand order repaired and checked by reading, not compiled, not run on hardware
//
#include <hip/hip_runtime.h>

#define BB 16
#define TT 120
#define PP 96
#define DD 128
#define LL 2

typedef __attribute__((ext_vector_type(16))) __bf16 v16bf;
typedef __attribute__((ext_vector_type(8)))  __bf16 v8bf;
typedef __attribute__((ext_vector_type(8)))  float  v8f;

static __device__ __forceinline__ v16bf join8(v8bf lo, v8bf hi) {
    return __builtin_shufflevector(lo, hi, 0,1,2,3,4,5,6,7,8,9,10,11,12,13,14,15);
}

// ---------------------------------------------------------------------------
// Pass 1: global distance statistics (count / sum / sumsq over masked d > 0)
// ---------------------------------------------------------------------------
__global__ void stats_kernel(const float* __restrict__ dist,
                             const unsigned char* __restrict__ mask,
                             double* __restrict__ stats)
{
    const long long total = (long long)BB * TT * PP * PP;
    double c = 0.0, s = 0.0, ss = 0.0;
    for (long long idx = (long long)blockIdx.x * blockDim.x + threadIdx.x;
         idx < total; idx += (long long)gridDim.x * blockDim.x) {
        const int q = (int)(idx % PP);
        const int p = (int)((idx / PP) % PP);
        const int b = (int)(idx / ((long long)TT * PP * PP));
        if (mask[b * PP + p] && mask[b * PP + q]) {
            const float d = dist[idx];
            if (d > 0.f) { c += 1.0; s += (double)d; ss += (double)d * (double)d; }
        }
    }
    #pragma unroll
    for (int m = 1; m <= 16; m <<= 1) {
        c  += __shfl_xor(c,  m, 32);
        s  += __shfl_xor(s,  m, 32);
        ss += __shfl_xor(ss, m, 32);
    }
    if ((threadIdx.x & 31) == 0) {
        atomicAdd(&stats[0], c);   // n
        atomicAdd(&stats[1], s);   // sum
        atomicAdd(&stats[2], ss);  // sum of squares
    }
}

// ---------------------------------------------------------------------------
// W (L,D,D) f32 -> Wt (L,E,D) bf16 so WMMA B-fragments are lane-contiguous
// ---------------------------------------------------------------------------
__global__ void wt_kernel(const float* __restrict__ W, __bf16* __restrict__ Wt)
{
    const int idx = blockIdx.x * blockDim.x + threadIdx.x;
    if (idx >= LL * DD * DD) return;
    const int l = idx / (DD * DD);
    const int r = idx - l * DD * DD;
    const int e = r / DD;
    const int d = r - e * DD;
    Wt[idx] = (__bf16)W[l * DD * DD + d * DD + e];
}

// ---------------------------------------------------------------------------
// Pass 2: per (b,t) normalized adjacency, written as bf16 row-major (p,q)
// ---------------------------------------------------------------------------
__global__ __launch_bounds__(256)
void adj_kernel(const float* __restrict__ dist,
                const unsigned char* __restrict__ mask,
                const double* __restrict__ stats,
                __bf16* __restrict__ adjOut)
{
    __shared__ float adjS[PP * PP];
    __shared__ float disS[PP];
    const int bt  = blockIdx.x;
    const int b   = bt / TT;
    const int tid = threadIdx.x;

    const double n    = stats[0];
    const double mean = stats[1] / n;
    const double var  = (stats[2] - n * mean * mean) / (n - 1.0);
    const float sigma  = (float)sqrt(var) + 1e-8f;
    const float inv2s2 = 1.f / (2.f * sigma * sigma);

    const float* dbt = dist + (size_t)bt * PP * PP;
    const unsigned char* mrow = mask + b * PP;

    for (int idx = tid; idx < PP * PP; idx += 256) {
        const int p = idx / PP, q = idx - p * PP;
        const bool vm = mrow[p] && mrow[q];
        const float d = dbt[idx];
        float a = vm ? __expf(-d * d * inv2s2) : 0.f;
        if (p == q && mrow[p]) a += 1.f;
        adjS[idx] = a;
    }
    __syncthreads();

    const int wv = tid >> 5, ln = tid & 31;
    for (int i = 0; i < 12; ++i) {               // 8 waves x 12 rows = 96
        const int p = wv * 12 + i;
        float sm = adjS[p * PP + ln] + adjS[p * PP + ln + 32] + adjS[p * PP + ln + 64];
        #pragma unroll
        for (int m = 1; m <= 16; m <<= 1) sm += __shfl_xor(sm, m, 32);
        float deg = fmaxf(sm, 1e-8f);
        deg = fmaxf(deg * (mrow[p] ? 1.f : 0.f), 1e-8f);
        if (ln == 0) disS[p] = rsqrtf(deg);
    }
    __syncthreads();

    for (int idx = tid; idx < PP * PP; idx += 256) {
        const int p = idx / PP, q = idx - p * PP;
        const bool vm = mrow[p] && mrow[q];
        const float v = vm ? disS[p] * adjS[idx] * disS[q] : 0.f;
        adjOut[(size_t)bt * PP * PP + idx] = (__bf16)v;
    }
}

// ---------------------------------------------------------------------------
// Pass 3: one GCN layer, fused: h = Xt*W ; o = A*h ; relu+bias+mask ;
// LayerNorm over D ; residual. One block per (b,t), 6 waves, wave w owns
// output rows [16w, 16w+16). All matrix math via v_wmma_f32_16x16x32_bf16.
// ---------------------------------------------------------------------------
__global__ __launch_bounds__(192)
void layer_kernel(const float* __restrict__ xin, float* __restrict__ xout,
                  const __bf16* __restrict__ adjW, const __bf16* __restrict__ Wt,
                  const float* __restrict__ bias, const float* __restrict__ gamma,
                  const float* __restrict__ beta, const unsigned char* __restrict__ mask)
{
    __shared__ __bf16 htL[DD * PP];   // h transposed: ht[e][q], 24 KB

    const int bt   = blockIdx.x;
    const int b    = bt / TT;
    const int t    = bt - b * TT;
    const int tid  = threadIdx.x;
    const int wv   = tid >> 5;
    const int lane = tid & 31;
    const int rr   = lane & 15;
    const int half = lane >> 4;
    const int m_off = wv * 16;
    const int row   = m_off + rr;     // A-matrix row held by this lane

    // ---- GEMM1 A-fragments: Xt[p][d] gathered from x(B,P,D,T), L2-resident
    const float* xrow = xin + ((size_t)(b * PP + row) * DD) * TT + t;
    v16bf aX[4];
    #pragma unroll
    for (int k = 0; k < 4; ++k) {
        const int k0 = k * 32 + half * 8;
        v16bf a;
        #pragma unroll
        for (int j = 0; j < 8; ++j) a[j]     = (__bf16)xrow[(size_t)(k0 + j) * TT];
        #pragma unroll
        for (int j = 0; j < 8; ++j) a[8 + j] = (__bf16)xrow[(size_t)(k0 + 16 + j) * TT];
        aX[k] = a;
    }

    // ---- GEMM1: h = Xt * W  (M=16 rows/wave, N=128, K=128) -> ht[e][q] LDS
    #pragma unroll
    for (int nb = 0; nb < 8; ++nb) {
        v8f c = {0.f,0.f,0.f,0.f,0.f,0.f,0.f,0.f};
        const int n = nb * 16 + rr;
        #pragma unroll
        for (int k = 0; k < 4; ++k) {
            const v16bf wf = join8(*(const v8bf*)(Wt + n * DD + k * 32 + half * 8), *(const v8bf*)(Wt + n * DD + k * 32 + 16 + half * 8));
            c = __builtin_amdgcn_wmma_f32_16x16x32_bf16(false, aX[k], false, wf,
                                                        (short)0, c, false, false);
        }
        v8bf hb;
        #pragma unroll
        for (int r = 0; r < 8; ++r) hb[r] = (__bf16)c[r];
        *(v8bf*)(&htL[n * PP + m_off + half * 8]) = hb;   // rows m_off+half*8+r
    }
    __syncthreads();

    // ---- GEMM2: o = A * h  (M=16 rows/wave, N=128, K=96)
    const __bf16* adjbt = adjW + (size_t)bt * PP * PP;
    v16bf a2[3];
    #pragma unroll
    for (int k = 0; k < 3; ++k) {
        const __bf16* ar = adjbt + row * PP + k * 32 + half * 8;
        a2[k] = join8(*(const v8bf*)ar, *(const v8bf*)(ar + 16));
    }
    v8f o[8];
    #pragma unroll
    for (int nb = 0; nb < 8; ++nb) {
        v8f c = {0.f,0.f,0.f,0.f,0.f,0.f,0.f,0.f};
        const int n = nb * 16 + rr;
        #pragma unroll
        for (int k = 0; k < 3; ++k) {
            const v16bf hf = join8(*(const v8bf*)(&htL[n * PP + k * 32 + half * 8]), *(const v8bf*)(&htL[n * PP + k * 32 + 16 + half * 8]));
            c = __builtin_amdgcn_wmma_f32_16x16x32_bf16(false, a2[k], false, hf,
                                                        (short)0, c, false, false);
        }
        o[nb] = c;
    }

    // ---- Epilogue: bias + relu + mask, LayerNorm over D (cross-lane), residual
    const unsigned char* mrow = mask + b * PP;
    const int p0 = m_off + half * 8;      // D-fragment row base for this lane
    float mk[8];
    #pragma unroll
    for (int r = 0; r < 8; ++r) mk[r] = mrow[p0 + r] ? 1.f : 0.f;
    float gv[8], bv[8], be[8];
    #pragma unroll
    for (int nb = 0; nb < 8; ++nb) {
        const int e = nb * 16 + rr;
        gv[nb] = gamma[e]; bv[nb] = beta[e]; be[nb] = bias[e];
    }
    #pragma unroll
    for (int nb = 0; nb < 8; ++nb)
        #pragma unroll
        for (int r = 0; r < 8; ++r)
            o[nb][r] = fmaxf(o[nb][r] + be[nb], 0.f) * mk[r];

    #pragma unroll
    for (int r = 0; r < 8; ++r) {
        float s1 = 0.f, s2 = 0.f;
        #pragma unroll
        for (int nb = 0; nb < 8; ++nb) { const float v = o[nb][r]; s1 += v; s2 += v * v; }
        #pragma unroll
        for (int m = 1; m <= 8; m <<= 1) {     // reduce over the 16 N-lanes
            s1 += __shfl_xor(s1, m, 32);
            s2 += __shfl_xor(s2, m, 32);
        }
        const float mu  = s1 * (1.f / 128.f);
        const float var = s2 * (1.f / 128.f) - mu * mu;
        const float rs  = rsqrtf(var + 1e-5f);
        const int p = p0 + r;
        const float* xr  = xin  + ((size_t)(b * PP + p) * DD) * TT + t;
        float*       orw = xout + ((size_t)(b * PP + p) * DD) * TT + t;
        #pragma unroll
        for (int nb = 0; nb < 8; ++nb) {
            const int e = nb * 16 + rr;
            const float v = (o[nb][r] - mu) * rs * gv[nb] + bv[nb];
            orw[(size_t)e * TT] = xr[(size_t)e * TT] + v;
        }
    }
}

// ---------------------------------------------------------------------------
extern "C" void kernel_launch(void* const* d_in, const int* in_sizes, int n_in,
                              void* d_out, int out_size, void* d_ws, size_t ws_size,
                              hipStream_t stream)
{
    const float* x            = (const float*)d_in[0];
    const float* dist         = (const float*)d_in[1];
    const unsigned char* mask = (const unsigned char*)d_in[2];   // bool (B,P)
    const float* W            = (const float*)d_in[3];
    const float* bias         = (const float*)d_in[4];
    const float* gamma        = (const float*)d_in[5];
    const float* beta         = (const float*)d_in[6];
    float* out = (float*)d_out;

    // workspace: [0,256) stats f64x3 ; [256, +64KB) Wt bf16 ; then adj bf16
    double* stats = (double*)d_ws;
    __bf16* Wt    = (__bf16*)((char*)d_ws + 256);
    __bf16* adjW  = (__bf16*)((char*)d_ws + 256 + (size_t)LL * DD * DD * 2);

    hipMemsetAsync(d_ws, 0, 256, stream);
    stats_kernel<<<512, 256, 0, stream>>>(dist, mask, stats);
    wt_kernel<<<(LL * DD * DD + 255) / 256, 256, 0, stream>>>(W, Wt);
    adj_kernel<<<BB * TT, 256, 0, stream>>>(dist, mask, stats, adjW);

    // layer 0: x -> out ; layer 1: out -> out (block-disjoint, in-place safe)
    layer_kernel<<<BB * TT, 192, 0, stream>>>(x,   out, adjW, Wt,
                                              bias, gamma, beta, mask);
    layer_kernel<<<BB * TT, 192, 0, stream>>>(out, out, adjW, Wt + DD * DD,
                                              bias + DD, gamma + DD, beta + DD, mask);
}
